// StickbreakingAttention_15006615732440
// MI455X (gfx1250) — matrix operand order repaired and checked by reading, not compiled, not run on hardware
//
#include <hip/hip_runtime.h>
#include <math.h>

// ---------------------------------------------------------------- constants
constexpr int   BB   = 2;
constexpr int   SS   = 1024;
constexpr int   HID  = 2048;
constexpr int   NH   = 16;
constexpr int   NKV  = 4;
constexpr int   HD   = 128;
constexpr int   QD   = NH * HD;    // 2048
constexpr int   KVD  = NKV * HD;   // 512
constexpr float SCALE = 0.08838834764831845f;   // 1/sqrt(128)
constexpr float EPSF  = 1e-6f;

// ---------------------------------------------------------------- types
typedef __attribute__((ext_vector_type(16))) __bf16         v16bf;
typedef __attribute__((ext_vector_type(8)))  float          v8f;
typedef __attribute__((ext_vector_type(16))) unsigned short v16u;
typedef __attribute__((ext_vector_type(8)))  unsigned short v8u;
typedef unsigned short u16;

static __device__ __forceinline__ u16 bf16_bits(float f) {
  union { float f; unsigned u; } c; c.f = f;
  return (u16)((c.u + 0x7FFFu + ((c.u >> 16) & 1u)) >> 16);  // RNE
}

static __device__ __forceinline__ v8f v8f_zero() {
  v8f z;
#pragma unroll
  for (int i = 0; i < 8; ++i) z[i] = 0.0f;
  return z;
}

// A fragment (16x32 bf16) from a bf16 row-major matrix [stride ld], tile (row0,k0).
// ISA 7.12.2: lane m=lane%16; kb=(lane<16?0:8); elems = [k0+kb ..+7] ++ [k0+kb+16 ..+7].
static __device__ __forceinline__ v16bf load_a_bf(const u16* base, int ld, int row0, int k0) {
  const int lane = threadIdx.x & 31;
  const int kb   = (lane < 16) ? 0 : 8;
  const u16* p = base + (size_t)(row0 + (lane & 15)) * ld + k0 + kb;
  v8u lo = *(const v8u*)(p);
  v8u hi = *(const v8u*)(p + 16);
  v16u u;
#pragma unroll
  for (int e = 0; e < 8; ++e) { u[e] = lo[e]; u[8 + e] = hi[e]; }
  return __builtin_bit_cast(v16bf, u);
}

// B fragment (32x16 bf16) where B = M^T, M bf16 row-major (N x K) [stride ld].
// Lane holds col n=lane%16; kb=(lane<16?0:8); elems = [k0+kb ..+7] ++ [k0+kb+16 ..+7].
static __device__ __forceinline__ v16bf load_bt_bf(const u16* base, int ld, int n0, int k0) {
  const int lane = threadIdx.x & 31;
  const int kb   = (lane < 16) ? 0 : 8;
  const u16* p = base + (size_t)(n0 + (lane & 15)) * ld + k0 + kb;
  v8u lo = *(const v8u*)(p);
  v8u hi = *(const v8u*)(p + 16);
  v16u u;
#pragma unroll
  for (int e = 0; e < 8; ++e) { u[e] = lo[e]; u[8 + e] = hi[e]; }
  return __builtin_bit_cast(v16bf, u);
}

static __device__ __forceinline__ float logsigf(float z) {
  // log(sigmoid(z)) = min(z,0) - log1p(exp(-|z|))   (numerically stable)
  return fminf(z, 0.0f) - log1pf(__expf(-fabsf(z)));
}

// ---------------------------------------------------------------- f32 -> bf16 bulk convert
__global__ void __launch_bounds__(256)
cvt_f32_bf16(const float4* __restrict__ src, u16* __restrict__ dst, int n4) {
  const int i = blockIdx.x * 256 + threadIdx.x;
  if (i < n4) {
    float4 f = src[i];
    union { u16 u[4]; uint2 v; } o;
    o.u[0] = bf16_bits(f.x); o.u[1] = bf16_bits(f.y);
    o.u[2] = bf16_bits(f.z); o.u[3] = bf16_bits(f.w);
    *(uint2*)(dst + (size_t)i * 4) = o.v;
  }
}

// ---------------------------------------------------------------- GEMM: C = X @ W^T  (bf16 in)
// X: (M x K) bf16 row-major, W: (N x K) bf16 row-major.
// One wave per block, 32x64 tile, software-pipelined K loop (register double buffering):
// the next k-block's 6 fragments are in flight while the current 8 WMMAs execute.
// OMODE 0: C = f32 row-major.  OMODE 2: C = bf16 scattered to vT[b][hkv][d][s].
template <int OMODE>
__global__ void __launch_bounds__(32)
gemm_bf16_wmma(const u16* __restrict__ X, const u16* __restrict__ W,
               void* __restrict__ Cout, int M, int N, int K) {
  const int row0 = blockIdx.x * 32;
  const int col0 = blockIdx.y * 64;
  if (row0 >= M || col0 >= N) return;   // uniform per wave; EXEC stays all-ones

  v8f acc[2][4];
#pragma unroll
  for (int s2 = 0; s2 < 2; ++s2)
#pragma unroll
    for (int t = 0; t < 4; ++t) acc[s2][t] = v8f_zero();

  // prologue: fragments for k-block 0
  v16bf a0 = load_a_bf(X, K, row0,      0);
  v16bf a1 = load_a_bf(X, K, row0 + 16, 0);
  v16bf b0 = load_bt_bf(W, K, col0 +  0, 0);
  v16bf b1 = load_bt_bf(W, K, col0 + 16, 0);
  v16bf b2 = load_bt_bf(W, K, col0 + 32, 0);
  v16bf b3 = load_bt_bf(W, K, col0 + 48, 0);

  for (int k0 = 0; k0 < K; k0 += 32) {
    // issue next k-block's loads first (clamped on last iter: harmless reload, no branch)
    const int kn = (k0 + 32 < K) ? (k0 + 32) : k0;
    v16bf na0 = load_a_bf(X, K, row0,      kn);
    v16bf na1 = load_a_bf(X, K, row0 + 16, kn);
    v16bf nb0 = load_bt_bf(W, K, col0 +  0, kn);
    v16bf nb1 = load_bt_bf(W, K, col0 + 16, kn);
    v16bf nb2 = load_bt_bf(W, K, col0 + 32, kn);
    v16bf nb3 = load_bt_bf(W, K, col0 + 48, kn);

    acc[0][0] = __builtin_amdgcn_wmma_f32_16x16x32_bf16(false, a0, false, b0, (short)0, acc[0][0], false, false);
    acc[1][0] = __builtin_amdgcn_wmma_f32_16x16x32_bf16(false, a1, false, b0, (short)0, acc[1][0], false, false);
    acc[0][1] = __builtin_amdgcn_wmma_f32_16x16x32_bf16(false, a0, false, b1, (short)0, acc[0][1], false, false);
    acc[1][1] = __builtin_amdgcn_wmma_f32_16x16x32_bf16(false, a1, false, b1, (short)0, acc[1][1], false, false);
    acc[0][2] = __builtin_amdgcn_wmma_f32_16x16x32_bf16(false, a0, false, b2, (short)0, acc[0][2], false, false);
    acc[1][2] = __builtin_amdgcn_wmma_f32_16x16x32_bf16(false, a1, false, b2, (short)0, acc[1][2], false, false);
    acc[0][3] = __builtin_amdgcn_wmma_f32_16x16x32_bf16(false, a0, false, b3, (short)0, acc[0][3], false, false);
    acc[1][3] = __builtin_amdgcn_wmma_f32_16x16x32_bf16(false, a1, false, b3, (short)0, acc[1][3], false, false);

    a0 = na0; a1 = na1; b0 = nb0; b1 = nb1; b2 = nb2; b3 = nb3;
  }

  const int lane = threadIdx.x & 31;
  const int mlo  = (lane >> 4) * 8;     // C/D layout: VGPR r -> row r + 8*(lane/16)
  const int nn   = lane & 15;
#pragma unroll
  for (int s2 = 0; s2 < 2; ++s2)
#pragma unroll
    for (int t = 0; t < 4; ++t)
#pragma unroll
      for (int r = 0; r < 8; ++r) {
        const int m = row0 + s2 * 16 + mlo + r;
        const int n = col0 + t * 16 + nn;
        if (OMODE == 0) {
          ((float*)Cout)[(size_t)m * N + n] = acc[s2][t][r];
        } else {  // OMODE == 2: V output, bf16, transposed to (b, hkv, d, s)
          const int b = m / SS, s = m % SS;
          const int hkv = n / HD, d = n % HD;
          ((u16*)Cout)[(((size_t)b * NKV + hkv) * HD + d) * SS + s] = bf16_bits(acc[s2][t][r]);
        }
      }
}

// ---------------------------------------------------------------- RoPE + RMS norm
// Reads f32 q/k from projection GEMMs, writes WMMA-ready bf16 q/k.
__global__ void __launch_bounds__(128)
rope_rmsnorm(const float* __restrict__ q32, const float* __restrict__ k32,
             const float* __restrict__ qw, const float* __restrict__ kw,
             u16* __restrict__ qbf, u16* __restrict__ kbf) {
  __shared__ float red[4];
  const int d  = threadIdx.x;                 // 0..127
  const int hh = blockIdx.x % (NH + NKV);
  const int s  = (blockIdx.x / (NH + NKV)) % SS;
  const int b  = blockIdx.x / ((NH + NKV) * SS);

  const float* vec; const float* w; u16* out;
  if (hh < NH) {
    const size_t off = ((size_t)(b * SS + s) * NH + hh) * HD;
    vec = q32 + off; out = qbf + off; w = qw;
  } else {
    const size_t off = ((size_t)(b * SS + s) * NKV + (hh - NH)) * HD;
    vec = k32 + off; out = kbf + off; w = kw;
  }

  const int   d2  = d & 63;
  const float inv = __expf(-(float)d2 * (9.210340371976184f / 64.0f)); // 10000^(-2*d2/128)
  const float ang = (float)s * inv;
  float sn, cs;
  __sincosf(ang, &sn, &cs);

  const float x1  = vec[d2];
  const float x2  = vec[d2 + 64];
  const float val = (d < 64) ? (x1 * cs - x2 * sn) : (x1 * sn + x2 * cs);

  float ssq = val * val;                      // mean of squares over HD
#pragma unroll
  for (int off = 16; off > 0; off >>= 1) ssq += __shfl_xor(ssq, off, 32);
  if ((d & 31) == 0) red[d >> 5] = ssq;
  __syncthreads();
  const float tot   = red[0] + red[1] + red[2] + red[3];
  const float scale = rsqrtf(tot * (1.0f / (float)HD) + EPSF);
  out[d] = bf16_bits(val * scale * w[d]);
}

// ---------------------------------------------------------------- Stick-breaking attention
// One wave per (b, h, 16-query tile). q/k bf16 row-major, v bf16 transposed (b,hkv,d,s).
// Key tiles of 32 walked high->low: reverse cumsum of log(1-sigmoid) becomes a per-tile
// 16-lane suffix scan (shfl in the C/D layout) + per-row f32 carry.
// V fragments are loaded BEFORE the scan so their latency hides under the VALU work.
__global__ void __launch_bounds__(32)
stickbreak_attn(const u16* __restrict__ q, const u16* __restrict__ k,
                const u16* __restrict__ vT, u16* __restrict__ o) {
  __shared__ u16 satt[16][40];                // bf16 att tile, padded rows (80B stride)

  const int ntq = SS / 16;
  const int tq  = blockIdx.x % ntq;
  const int h   = (blockIdx.x / ntq) % NH;
  const int b   = blockIdx.x / (ntq * NH);
  const int hkv = h / (NH / NKV);
  const int i0  = tq * 16;

  const u16* qb = q  + (size_t)b * SS * QD  + (size_t)h   * HD;
  const u16* kb = k  + (size_t)b * SS * KVD + (size_t)hkv * HD;
  const u16* vb = vT + ((size_t)b * NKV + hkv) * HD * SS;   // (HD x S) for this (b,hkv)
  u16*       ob = o  + (size_t)b * SS * QD  + (size_t)h   * HD;

  const int lane = threadIdx.x & 31;
  const int half = lane >> 4;
  const int nn   = lane & 15;

  v16bf qa[4];                                // Q fragments (16 x 128), loaded once
#pragma unroll
  for (int c = 0; c < 4; ++c) qa[c] = load_a_bf(qb, QD, i0, c * 32);

  v8f oacc[8];                                // O accumulator, 16 x 128 f32
#pragma unroll
  for (int t = 0; t < 8; ++t) oacc[t] = v8f_zero();
  float carry[8];                             // per-row running sum of masked log(1-p)
#pragma unroll
  for (int r = 0; r < 8; ++r) carry[r] = 0.0f;

  const int jt_hi = (i0 + 14) >> 5;           // last key tile with any j <= i0+14
  for (int jt = jt_hi; jt >= 0; --jt) {
    const int j0 = jt * 32;

    // prefetch next tile's K rows into near caches (uniform-guarded, L2-resident data)
    if (jt > 0) {
      __builtin_prefetch(kb + (size_t)(j0 - 32 + lane) * KVD, 0, 1);
      __builtin_prefetch(kb + (size_t)(j0 - 32 + lane) * KVD + 128, 0, 1);
    }

    // ---- load all K fragments, then run the 8 QK^T WMMAs
    v16bf bk[8];
#pragma unroll
    for (int c = 0; c < 4; ++c) {
      bk[2 * c]     = load_bt_bf(kb, KVD, j0,      c * 32);
      bk[2 * c + 1] = load_bt_bf(kb, KVD, j0 + 16, c * 32);
    }
    v8f c0 = v8f_zero(), c1 = v8f_zero();
#pragma unroll
    for (int c = 0; c < 4; ++c) {
      c0 = __builtin_amdgcn_wmma_f32_16x16x32_bf16(false, qa[c], false, bk[2 * c],     (short)0, c0, false, false);
      c1 = __builtin_amdgcn_wmma_f32_16x16x32_bf16(false, qa[c], false, bk[2 * c + 1], (short)0, c1, false, false);
    }

    // ---- V fragments for this tile: independent of the logits; issue before the scan
    v16bf bv[8];
#pragma unroll
    for (int t = 0; t < 8; ++t) bv[t] = load_bt_bf(vb, SS, t * 16, j0);

    // ---- stick-breaking weights + reverse suffix scan per query row.
    // D-layout: VGPR r, 16-lane group holds row (i0 + 8*half + r), cols j0+nn (+16).
#pragma unroll
    for (int r = 0; r < 8; ++r) {
      const int irow = i0 + half * 8 + r;
      const int ja = j0 + nn, jb2 = ja + 16;
      const float z0 = c0[r] * SCALE, z1 = c1[r] * SCALE;
      const float lp0 = logsigf(z0), lp1 = logsigf(z1);
      float w0 = (ja  < irow) ? logsigf(-z0) : 0.0f;
      float w1 = (jb2 < irow) ? logsigf(-z1) : 0.0f;
      float inc0 = w0, inc1 = w1;               // inclusive suffix sums within 16 lanes
#pragma unroll
      for (int off = 1; off < 16; off <<= 1) {
        const float t0 = __shfl_down(inc0, off, 16);
        const float t1 = __shfl_down(inc1, off, 16);
        if (nn + off < 16) { inc0 += t0; inc1 += t1; }
      }
      const float tot0 = __shfl(inc0, 0, 16);   // tile-half totals (broadcast)
      const float tot1 = __shfl(inc1, 0, 16);
      const float cum1 = (inc1 - w1) + carry[r];
      const float cum0 = (inc0 - w0) + tot1 + carry[r];
      const float a0 = (ja  < irow) ? __expf(lp0 + cum0) : 0.0f;
      const float a1 = (jb2 < irow) ? __expf(lp1 + cum1) : 0.0f;
      carry[r] += tot0 + tot1;
      satt[half * 8 + r][nn]      = bf16_bits(a0);  // relay through LDS to A-layout
      satt[half * 8 + r][nn + 16] = bf16_bits(a1);
    }

    // ---- O += att @ V  (A = 16x32 att from LDS; V fragments already resident)
    v16bf pa = load_a_bf(&satt[0][0], 40, 0, 0);
#pragma unroll
    for (int t = 0; t < 8; ++t)
      oacc[t] = __builtin_amdgcn_wmma_f32_16x16x32_bf16(false, pa, false, bv[t], (short)0, oacc[t], false, false);
  }

#pragma unroll
  for (int t = 0; t < 8; ++t)
#pragma unroll
    for (int r = 0; r < 8; ++r)
      ob[(size_t)(i0 + half * 8 + r) * QD + t * 16 + nn] = bf16_bits(oacc[t][r]);
}

// ---------------------------------------------------------------- launch
extern "C" void kernel_launch(void* const* d_in, const int* in_sizes, int n_in,
                              void* d_out, int out_size, void* d_ws, size_t ws_size,
                              hipStream_t stream) {
  (void)in_sizes; (void)n_in; (void)out_size; (void)ws_size;
  const float* hs = (const float*)d_in[0];
  const float* Wq = (const float*)d_in[1];
  const float* Wk = (const float*)d_in[2];
  const float* Wv = (const float*)d_in[3];
  const float* Wo = (const float*)d_in[4];
  const float* qw = (const float*)d_in[5];
  const float* kw = (const float*)d_in[6];
  float* out = (float*)d_out;

  // scratch layout (bump allocator; ~68 MB total, L2-resident)
  char* wsp = (char*)d_ws;
  auto alloc = [&](size_t bytes) -> char* {
    char* p = wsp; wsp += (bytes + 255) & ~(size_t)255; return p;
  };
  const size_t nHS = (size_t)BB * SS * HID;   // 4M
  const size_t nWq = (size_t)QD * HID;        // 4M
  const size_t nWk = (size_t)KVD * HID;       // 1M
  const size_t nQ  = (size_t)BB * SS * QD;    // 4M
  const size_t nK  = (size_t)BB * SS * KVD;   // 1M

  u16*   hs_bf = (u16*)alloc(nHS * 2);
  u16*   wq_bf = (u16*)alloc(nWq * 2);
  u16*   wk_bf = (u16*)alloc(nWk * 2);
  u16*   wv_bf = (u16*)alloc(nWk * 2);
  u16*   wo_bf = (u16*)alloc(nWq * 2);
  float* q32   = (float*)alloc(nQ * 4);
  float* k32   = (float*)alloc(nK * 4);
  u16*   q_bf  = (u16*)alloc(nQ * 2);
  u16*   k_bf  = (u16*)alloc(nK * 2);
  u16*   vT_bf = (u16*)alloc(nK * 2);
  u16*   oat   = (u16*)alloc(nQ * 2);

  // 1) one-time f32 -> bf16 conversion of activations + weights
  cvt_f32_bf16<<<(int)(nHS / 1024), 256, 0, stream>>>((const float4*)hs, hs_bf, (int)(nHS / 4));
  cvt_f32_bf16<<<(int)(nWq / 1024), 256, 0, stream>>>((const float4*)Wq, wq_bf, (int)(nWq / 4));
  cvt_f32_bf16<<<(int)(nWk / 1024), 256, 0, stream>>>((const float4*)Wk, wk_bf, (int)(nWk / 4));
  cvt_f32_bf16<<<(int)(nWk / 1024), 256, 0, stream>>>((const float4*)Wv, wv_bf, (int)(nWk / 4));
  cvt_f32_bf16<<<(int)(nWq / 1024), 256, 0, stream>>>((const float4*)Wo, wo_bf, (int)(nWq / 4));

  const int M = BB * SS;                      // 2048 token rows
  // 2) projections (V goes straight to bf16, transposed per (b,kv-head))
  gemm_bf16_wmma<0><<<dim3(M / 32, QD  / 64), 32, 0, stream>>>(hs_bf, wq_bf, q32,   M, QD,  HID);
  gemm_bf16_wmma<0><<<dim3(M / 32, KVD / 64), 32, 0, stream>>>(hs_bf, wk_bf, k32,   M, KVD, HID);
  gemm_bf16_wmma<2><<<dim3(M / 32, KVD / 64), 32, 0, stream>>>(hs_bf, wv_bf, vT_bf, M, KVD, HID);
  // 3) RoPE + RMS norm, f32 in -> bf16 out
  rope_rmsnorm<<<BB * SS * (NH + NKV), 128, 0, stream>>>(q32, k32, qw, kw, q_bf, k_bf);
  // 4) fused stick-breaking attention
  stickbreak_attn<<<BB * NH * (SS / 16), 32, 0, stream>>>(q_bf, k_bf, vT_bf, oat);
  // 5) output projection -> f32
  gemm_bf16_wmma<0><<<dim3(M / 32, HID / 64), 32, 0, stream>>>(oat, wo_bf, out, M, HID, HID);
}
